// BKG_encoder_1357209665643
// MI455X (gfx1250) — matrix operand order repaired and checked by reading, not compiled, not run on hardware
//
#include <hip/hip_runtime.h>
#include <math.h>

// ---------------- problem constants (match reference) ----------------
#define NN      8192          // nodes
#define HIDD    256           // hidden
#define NHEADS  8
#define HDIM    32            // head dim
#define DEG     16            // out-edges per node (contiguous per node)
#define EE      (NN * DEG)    // edges = 131072
#define BB      128           // graphs
#define NPER    64            // nodes per graph
#define BN_EPS  1e-5f
#define QSCALE  0.17677669529663687f   // 32^-0.5

typedef __attribute__((ext_vector_type(16))) __bf16 v16bf;
typedef __attribute__((ext_vector_type(8)))  __bf16 v8bf;
typedef __attribute__((ext_vector_type(8)))  float  v8f;

__device__ __forceinline__ __bf16 f2bf(float f) {
  union { float f; unsigned u; } in; in.f = f;
  unsigned r = (in.u + 0x7FFFu + ((in.u >> 16) & 1u)) >> 16;   // RNE
  union { unsigned short s; __bf16 b; } out; out.s = (unsigned short)r;
  return out.b;
}

__device__ __forceinline__ float gelu_exact(float v) {
  return 0.5f * v * (1.0f + erff(v * 0.70710678118654752f));
}

// ---------------- WMMA bf16 GEMM:  C = epi(alpha*(A@B + bias)) [+ resid] ----
// A: M x K row-major bf16.  Bt: N x K row-major bf16 (i.e. B transposed).
// Each wave owns a 16(M) x 64(N) output strip: one A fragment is reused by
// 4 WMMAs per 32-deep K step (4 accumulators). 8 waves per block.
// K multiple of 32, M multiple of 16, N multiple of 64 (always true here).
__global__ __launch_bounds__(256)
void bkg_wmma_gemm(const __bf16* __restrict__ A, const __bf16* __restrict__ Bt,
                   const float* __restrict__ bias, const float* __restrict__ resid,
                   float* __restrict__ C, int M, int Nn, int K,
                   float alpha, int do_gelu) {
  const int lane = threadIdx.x & 31;
  const int wave = threadIdx.x >> 5;
  const int ntN4 = Nn >> 6;                       // 64-wide supertiles along N
  const int tiles = (M >> 4) * ntN4;
  const int tile = blockIdx.x * 8 + wave;
  if (tile >= tiles) return;                      // wave-uniform: EXEC stays full
  const int tm  = (tile / ntN4) << 4;
  const int tn0 = (tile % ntN4) << 6;
  const int mn = lane & 15;                       // row of A-frag / row of Bt-frag
  const int hi = lane >> 4;                       // lane-group

  const __bf16* __restrict__ arow = A  + (size_t)(tm + mn) * K;
  const __bf16* __restrict__ brow = Bt + (size_t)(tn0 + mn) * K;
  const size_t bstep = (size_t)16 * K;            // next 16-col group of Bt

  v8f acc[4] = {v8f{}, v8f{}, v8f{}, v8f{}};
  #pragma unroll 2
  for (int k0 = 0; k0 < K; k0 += 32) {
    // A 16x32 bf16 fragment: elem e -> K = k0+hi*16+e
    union { v16bf v; v8bf h[2]; } af;
    af.h[0] = *(const v8bf*)(arow + k0 + hi * 16);
    af.h[1] = *(const v8bf*)(arow + k0 + hi * 16 + 8);
    // B 32x16 bf16 fragments (transposed weights): elem e -> K = k0+hi*16+e
    const int bo = k0 + hi * 16;
    v16bf bf0 = *(const v16bf*)(brow + bo);
    v16bf bf1 = *(const v16bf*)(brow + bstep + bo);
    v16bf bf2 = *(const v16bf*)(brow + 2 * bstep + bo);
    v16bf bf3 = *(const v16bf*)(brow + 3 * bstep + bo);
    acc[0] = __builtin_amdgcn_wmma_f32_16x16x32_bf16(false, af.v, false, bf0,
                                                     (short)0, acc[0], false, false);
    acc[1] = __builtin_amdgcn_wmma_f32_16x16x32_bf16(false, af.v, false, bf1,
                                                     (short)0, acc[1], false, false);
    acc[2] = __builtin_amdgcn_wmma_f32_16x16x32_bf16(false, af.v, false, bf2,
                                                     (short)0, acc[2], false, false);
    acc[3] = __builtin_amdgcn_wmma_f32_16x16x32_bf16(false, af.v, false, bf3,
                                                     (short)0, acc[3], false, false);
  }

  #pragma unroll
  for (int sub = 0; sub < 4; ++sub) {
    const int tn = tn0 + sub * 16;
    const float bv = bias ? bias[tn + mn] : 0.0f;
    #pragma unroll
    for (int r = 0; r < 8; ++r) {                 // D vgpr r -> row tm + r + 8*hi
      const int rowi = tm + r + 8 * hi;
      const size_t idx = (size_t)rowi * Nn + tn + mn;
      float v = (acc[sub][r] + bv) * alpha;
      if (do_gelu) v = gelu_exact(v);
      if (resid)  v += resid[idx];
      C[idx] = v;
    }
  }
}

// ---------------- conversions ----------------
__global__ __launch_bounds__(256)
void bkg_cvt_bf16(const float* __restrict__ x, __bf16* __restrict__ y, int n) {
  int i = blockIdx.x * 256 + threadIdx.x;
  if (i < n) y[i] = f2bf(x[i]);
}

__global__ __launch_bounds__(256)
void bkg_transpose_bf16(const float* __restrict__ W, __bf16* __restrict__ Wt,
                        int K, int Nn) {   // W: K x N -> Wt: N x K
  int t = blockIdx.x * 256 + threadIdx.x;
  if (t >= K * Nn) return;
  int k = t / Nn, n = t - k * Nn;
  Wt[(size_t)n * K + k] = f2bf(W[t]);
}

// ---------------- sparse attention ----------------
// scores[e, hd] = sum_d q[row_e, d*8+hd] * k[col_e, d*8+hd]   (q pre-scaled)
__global__ __launch_bounds__(256)
void bkg_sddmm(const float* __restrict__ q, const float* __restrict__ k,
               const int* __restrict__ row, const int* __restrict__ col,
               float* __restrict__ scores) {
  int e = blockIdx.x * 256 + threadIdx.x;
  if (e >= EE) return;
  const int r = row[e], c = col[e];
  const float4* qr = (const float4*)(q + (size_t)r * HIDD);
  const float4* kc = (const float4*)(k + (size_t)c * HIDD);
  float acc[8] = {0, 0, 0, 0, 0, 0, 0, 0};
  #pragma unroll 8
  for (int i = 0; i < HIDD / 4; ++i) {
    float4 a = qr[i], b = kc[i];
    const int h0 = (i & 1) * 4;                  // channel = 4i.. -> head = channel & 7
    acc[h0 + 0] += a.x * b.x; acc[h0 + 1] += a.y * b.y;
    acc[h0 + 2] += a.z * b.z; acc[h0 + 3] += a.w * b.w;
  }
  float* s = scores + (size_t)e * NHEADS;
  #pragma unroll
  for (int hd = 0; hd < NHEADS; ++hd) s[hd] = acc[hd];
}

// row-wise softmax over each node's 16 contiguous edges, per head (in place)
__global__ __launch_bounds__(256)
void bkg_edge_softmax(float* __restrict__ scores) {
  int t = blockIdx.x * 256 + threadIdx.x;       // (node, head)
  if (t >= NN * NHEADS) return;
  const int n = t >> 3, hd = t & 7;
  float sc[DEG];
  #pragma unroll
  for (int j = 0; j < DEG; ++j) sc[j] = scores[(size_t)(n * DEG + j) * NHEADS + hd];
  float mx = sc[0];
  #pragma unroll
  for (int j = 1; j < DEG; ++j) mx = fmaxf(mx, sc[j]);
  float s = 0.f;
  #pragma unroll
  for (int j = 0; j < DEG; ++j) { sc[j] = expf(sc[j] - mx); s += sc[j]; }
  const float inv = 1.0f / s;
  #pragma unroll
  for (int j = 0; j < DEG; ++j) scores[(size_t)(n * DEG + j) * NHEADS + hd] = sc[j] * inv;
}

// out[n, c] = sum_j attn[e_j, c&7] * v[col_j, c]
__global__ __launch_bounds__(256)
void bkg_spmm(const float* __restrict__ attnw, const float* __restrict__ v,
              const int* __restrict__ col, float* __restrict__ out) {
  int t = blockIdx.x * 256 + threadIdx.x;
  if (t >= NN * HIDD) return;
  const int n = t >> 8, c = t & 255, hd = c & 7;
  float s = 0.f;
  #pragma unroll
  for (int j = 0; j < DEG; ++j) {
    const int e = n * DEG + j;
    s += attnw[(size_t)e * NHEADS + hd] * v[(size_t)col[e] * HIDD + c];
  }
  out[t] = s;
}

// ---------------- BatchNorm (training-mode, biased var) ----------------
__global__ __launch_bounds__(256)
void bkg_colstats(const float* __restrict__ X, float* __restrict__ stats, int M, int C) {
  const int c = blockIdx.x;
  float s = 0.f, s2 = 0.f;
  for (int n = threadIdx.x; n < M; n += 256) {
    float v = X[(size_t)n * C + c]; s += v; s2 += v * v;
  }
  __shared__ float sh[512];
  sh[threadIdx.x] = s; sh[256 + threadIdx.x] = s2; __syncthreads();
  for (int o = 128; o > 0; o >>= 1) {
    if (threadIdx.x < o) { sh[threadIdx.x] += sh[threadIdx.x + o];
                           sh[256 + threadIdx.x] += sh[256 + threadIdx.x + o]; }
    __syncthreads();
  }
  if (threadIdx.x == 0) {
    float mu = sh[0] / (float)M;
    stats[c] = mu;
    stats[C + c] = sh[256] / (float)M - mu * mu;
  }
}

__global__ __launch_bounds__(256)
void bkg_bn_apply(const float* __restrict__ x, const float* __restrict__ stats,
                  const float* __restrict__ g, const float* __restrict__ b,
                  float* __restrict__ y, int total, int C, int do_gelu) {
  int t = blockIdx.x * 256 + threadIdx.x;
  if (t >= total) return;
  const int c = t & (C - 1);
  float v = g[c] * (x[t] - stats[c]) * rsqrtf(stats[C + c] + BN_EPS) + b[c];
  if (do_gelu) v = gelu_exact(v);
  y[t] = v;
}

// ---------------- readout head ----------------
__global__ __launch_bounds__(256)
void bkg_gather_base(const float* __restrict__ h, const int* __restrict__ head_ids,
                     const int* __restrict__ tail_ids, __bf16* __restrict__ base) {
  int t = blockIdx.x * 256 + threadIdx.x;       // BB * 512
  if (t >= BB * 2 * HIDD) return;
  const int g = t >> 9, c = t & 511;
  const int node = (c < HIDD) ? head_ids[g] : tail_ids[g];
  base[t] = f2bf(h[(size_t)node * HIDD + (c & 255)]);
}

__global__ __launch_bounds__(256)
void bkg_node_attn(const float* __restrict__ hw, const float* __restrict__ tb,
                   const int* __restrict__ gid, float* __restrict__ nattn) {
  int n = blockIdx.x * 256 + threadIdx.x;
  if (n >= NN) return;
  const float* a = hw + (size_t)n * HIDD;
  const float* t = tb + (size_t)gid[n] * HIDD;
  float s = 0.f;
  #pragma unroll 4
  for (int c = 0; c < HIDD; ++c) s += a[c] * t[c];
  nattn[n] = s;
}

__global__ __launch_bounds__(64)
void bkg_graph_softmax(const float* __restrict__ nattn, float* __restrict__ a) {
  __shared__ float sh[NPER];
  const int g = blockIdx.x, i = threadIdx.x;
  const float v = nattn[g * NPER + i];
  sh[i] = v; __syncthreads();
  for (int o = 32; o > 0; o >>= 1) { if (i < o) sh[i] = fmaxf(sh[i], sh[i + o]); __syncthreads(); }
  const float mx = sh[0]; __syncthreads();
  const float ex = expf(v - mx);
  sh[i] = ex; __syncthreads();
  for (int o = 32; o > 0; o >>= 1) { if (i < o) sh[i] += sh[i + o]; __syncthreads(); }
  a[g * NPER + i] = ex / sh[0];
}

__global__ __launch_bounds__(256)
void bkg_readout(const float* __restrict__ h, const float* __restrict__ a,
                 float* __restrict__ out) {
  int t = blockIdx.x * 256 + threadIdx.x;       // BB * HIDD
  if (t >= BB * HIDD) return;
  const int g = t >> 8, c = t & 255;
  float s = 0.f;
  #pragma unroll 4
  for (int i = 0; i < NPER; ++i) {
    const int n = g * NPER + i;
    s += h[(size_t)n * HIDD + c] * a[n];
  }
  out[t] = s * (1.0f / (float)NPER);
}

__global__ __launch_bounds__(256)
void bkg_center_loss(const float* __restrict__ h, const int* __restrict__ head_ids,
                     const int* __restrict__ tail_ids, const int* __restrict__ pairs,
                     const float* __restrict__ centers, float* __restrict__ out) {
  __shared__ float sh[256];
  const int i = threadIdx.x;                    // 0..255 = 2*BB feats
  const int g = i & 127, part = i >> 7;
  const int node  = part ? tail_ids[g] : head_ids[g];
  const int label = pairs[g * 2 + part];
  const float* f = h + (size_t)node * HIDD;
  const float* c = centers + (size_t)label * HIDD;
  float d2 = 0.f;
  #pragma unroll 4
  for (int j = 0; j < HIDD; ++j) { float d = f[j] - c[j]; d2 += d * d; }
  d2 = fminf(fmaxf(d2, 1e-12f), 1e12f);
  sh[i] = d2; __syncthreads();
  for (int o = 128; o > 0; o >>= 1) { if (i < o) sh[i] += sh[i + o]; __syncthreads(); }
  if (i == 0) out[0] = sh[0] / 256.0f;          // /(2B)
}

// ---------------- host orchestration ----------------
static inline void launch_gemm(const __bf16* A, const __bf16* Bt, const float* bias,
                               const float* resid, float* C, int M, int Nn, int K,
                               float alpha, int gelu, hipStream_t s) {
  const int tiles = (M / 16) * (Nn / 64);       // 16x64 strip per wave
  const int blocks = (tiles + 7) / 8;
  bkg_wmma_gemm<<<blocks, 256, 0, s>>>(A, Bt, bias, resid, C, M, Nn, K, alpha, gelu);
}

extern "C" void kernel_launch(void* const* d_in, const int* in_sizes, int n_in,
                              void* d_out, int out_size, void* d_ws, size_t ws_size,
                              hipStream_t stream) {
  (void)in_sizes; (void)n_in; (void)out_size;
  // Input leaf order: jax pytree flatten (dict keys sorted at every level):
  // col, drug_pairs, graph_id, h, head_ids,
  //   params{ W_base, W_h, centers, layers[0..2]{bn1(g,b),bn2(g,b),f1(w,b),f2(w,b),k,o,q,v} },
  // row, tail_ids
  const int*   col      = (const int*)d_in[0];
  const int*   pairs    = (const int*)d_in[1];
  const int*   gid      = (const int*)d_in[2];
  const float* h_in     = (const float*)d_in[3];
  const int*   head_ids = (const int*)d_in[4];
  const float* W_base   = (const float*)d_in[5];
  const float* W_h      = (const float*)d_in[6];
  const float* centers  = (const float*)d_in[7];
  const int*   row      = (const int*)d_in[56];
  const int*   tail_ids = (const int*)d_in[57];
  auto P = [&](int L, int j) -> const float* { return (const float*)d_in[8 + 16 * L + j]; };
  // per-layer leaf offsets: 0 bn1_g 1 bn1_b 2 bn2_g 3 bn2_b 4 f1_w 5 f1_b
  //                         6 f2_w 7 f2_b 8 k_w 9 k_b 10 o_w 11 o_b 12 q_w 13 q_b 14 v_w 15 v_b

  // workspace layout (256B aligned)
  char* ws = (char*)d_ws;
  const size_t SZ_H   = (size_t)NN * HIDD * 4;           // 8 MB
  const size_t SZ_MID = (size_t)NN * 512 * 4;            // 16 MB
  const size_t SZ_SC  = (size_t)EE * NHEADS * 4;         // 4 MB
  size_t off = 0;
  auto take = [&](size_t bytes) { size_t o = off; off += (bytes + 255) & ~(size_t)255; return o; };
  float*  hbuf   = (float*)(ws + take(SZ_H));
  float*  xbuf   = (float*)(ws + take(SZ_H));
  float*  qbuf   = (float*)(ws + take(SZ_H));   // also hw in the head
  float*  kbuf   = (float*)(ws + take(SZ_H));   // also tb in the head
  float*  vbuf   = (float*)(ws + take(SZ_H));
  float*  midbuf = (float*)(ws + take(SZ_MID));
  float*  scores = (float*)(ws + take(SZ_SC));
  float*  nattn  = (float*)(ws + take((size_t)NN * 4));
  float*  asoft  = (float*)(ws + take((size_t)NN * 4));
  float*  stats  = (float*)(ws + take(4096));
  __bf16* abf    = (__bf16*)(ws + take((size_t)NN * 512 * 2));   // activation staging (max N x 512)
  __bf16* basebf = (__bf16*)(ws + take((size_t)BB * 512 * 2));
  __bf16* war    = (__bf16*)(ws + take((size_t)1769472 * 2));    // transposed weight arena
  if (ws_size < off) return;  // not enough scratch; bail deterministically

  const size_t WL = 524288;   // bf16 elems per layer in arena
  auto qwt = [&](int L) { return war + (size_t)L * WL + 0;      };
  auto kwt = [&](int L) { return war + (size_t)L * WL + 65536;  };
  auto vwt = [&](int L) { return war + (size_t)L * WL + 131072; };
  auto owt = [&](int L) { return war + (size_t)L * WL + 196608; };
  auto f1t = [&](int L) { return war + (size_t)L * WL + 262144; };  // 512 x 256
  auto f2t = [&](int L) { return war + (size_t)L * WL + 393216; };  // 256 x 512
  __bf16* wht = war + 3 * WL;            // 256 x 256
  __bf16* wbt = war + 3 * WL + 65536;    // 256 x 512

  // ---- prep: working copy of h + transpose/convert all weights to bf16 ----
  hipMemcpyAsync(hbuf, h_in, SZ_H, hipMemcpyDeviceToDevice, stream);
  for (int L = 0; L < 3; ++L) {
    bkg_transpose_bf16<<<256, 256, 0, stream>>>(P(L, 12), qwt(L), HIDD, HIDD);
    bkg_transpose_bf16<<<256, 256, 0, stream>>>(P(L, 8),  kwt(L), HIDD, HIDD);
    bkg_transpose_bf16<<<256, 256, 0, stream>>>(P(L, 14), vwt(L), HIDD, HIDD);
    bkg_transpose_bf16<<<256, 256, 0, stream>>>(P(L, 10), owt(L), HIDD, HIDD);
    bkg_transpose_bf16<<<512, 256, 0, stream>>>(P(L, 4),  f1t(L), HIDD, 512);
    bkg_transpose_bf16<<<512, 256, 0, stream>>>(P(L, 6),  f2t(L), 512, HIDD);
  }
  bkg_transpose_bf16<<<256, 256, 0, stream>>>(W_h,    wht, HIDD, HIDD);
  bkg_transpose_bf16<<<512, 256, 0, stream>>>(W_base, wbt, 512, HIDD);

  // ---- 3 encoder layers ----
  for (int L = 0; L < 3; ++L) {
    bkg_cvt_bf16<<<NN * HIDD / 256, 256, 0, stream>>>(hbuf, abf, NN * HIDD);
    launch_gemm(abf, qwt(L), P(L, 13), nullptr, qbuf, NN, HIDD, HIDD, QSCALE, 0, stream);
    launch_gemm(abf, kwt(L), P(L, 9),  nullptr, kbuf, NN, HIDD, HIDD, 1.0f,   0, stream);
    launch_gemm(abf, vwt(L), P(L, 15), nullptr, vbuf, NN, HIDD, HIDD, 1.0f,   0, stream);

    bkg_sddmm<<<EE / 256, 256, 0, stream>>>(qbuf, kbuf, row, col, scores);
    bkg_edge_softmax<<<NN * NHEADS / 256, 256, 0, stream>>>(scores);
    bkg_spmm<<<NN * HIDD / 256, 256, 0, stream>>>(scores, vbuf, col, xbuf);

    bkg_cvt_bf16<<<NN * HIDD / 256, 256, 0, stream>>>(xbuf, abf, NN * HIDD);
    launch_gemm(abf, owt(L), P(L, 11), /*resid=*/hbuf, xbuf, NN, HIDD, HIDD, 1.0f, 0, stream);

    bkg_colstats<<<HIDD, 256, 0, stream>>>(xbuf, stats, NN, HIDD);
    bkg_bn_apply<<<NN * HIDD / 256, 256, 0, stream>>>(xbuf, stats, P(L, 0), P(L, 1),
                                                      hbuf, NN * HIDD, HIDD, 1);
    bkg_cvt_bf16<<<NN * HIDD / 256, 256, 0, stream>>>(hbuf, abf, NN * HIDD);
    launch_gemm(abf, f1t(L), P(L, 5), nullptr, midbuf, NN, 512, HIDD, 1.0f, /*gelu=*/1, stream);
    bkg_cvt_bf16<<<NN * 512 / 256, 256, 0, stream>>>(midbuf, abf, NN * 512);
    launch_gemm(abf, f2t(L), P(L, 7), /*resid=*/hbuf, xbuf, NN, HIDD, 512, 1.0f, 0, stream);

    bkg_colstats<<<HIDD, 256, 0, stream>>>(xbuf, stats, NN, HIDD);
    bkg_bn_apply<<<NN * HIDD / 256, 256, 0, stream>>>(xbuf, stats, P(L, 2), P(L, 3),
                                                      hbuf, NN * HIDD, HIDD, 0);
  }

  // ---- readout head ----
  bkg_cvt_bf16<<<NN * HIDD / 256, 256, 0, stream>>>(hbuf, abf, NN * HIDD);
  launch_gemm(abf, wht, nullptr, nullptr, qbuf, NN, HIDD, HIDD, 1.0f, 0, stream);      // h @ W_h
  bkg_gather_base<<<BB * 512 / 256, 256, 0, stream>>>(hbuf, head_ids, tail_ids, basebf);
  launch_gemm(basebf, wbt, nullptr, nullptr, kbuf, BB, HIDD, 512, 1.0f, 0, stream);    // base @ W_base
  bkg_node_attn<<<NN / 256, 256, 0, stream>>>(qbuf, kbuf, gid, nattn);
  bkg_graph_softmax<<<BB, 64, 0, stream>>>(nattn, asoft);
  bkg_readout<<<BB * HIDD / 256, 256, 0, stream>>>(hbuf, asoft, (float*)d_out);
  bkg_center_loss<<<1, 256, 0, stream>>>(hbuf, head_ids, tail_ids, pairs, centers,
                                         (float*)d_out + BB * HIDD);
}
